// Sequence_22368189678306
// MI455X (gfx1250) — compile-verified, matrix operand order checked by reading, not run on hardware
//
#include <hip/hip_runtime.h>
#include <hip/hip_bf16.h>

typedef _Float16 v16h __attribute__((ext_vector_type(16)));
typedef float    v8f  __attribute__((ext_vector_type(8)));

#define HID   51
#define G4    204            // 4*HID gate columns
#define NT    13             // column tiles of 16 (204 -> 208)
#define WAVES 4
#define BLK   (WAVES * 32)

// Fast transcendentals on the CDNA5 trans pipe (v_exp_f32 / v_rcp_f32)
__device__ __forceinline__ float fexp (float x){ return __builtin_amdgcn_exp2f(x * 1.4426950408889634f); }
__device__ __forceinline__ float frcp (float x){ return __builtin_amdgcn_rcpf(x); }
__device__ __forceinline__ float fsig (float x){ return frcp(1.0f + fexp(-x)); }
__device__ __forceinline__ float ftanh_(float x){ return 1.0f - 2.0f * frcp(1.0f + fexp(2.0f * x)); }

__global__ __launch_bounds__(BLK) void fused_lstm_wmma(
    const float* __restrict__ x,
    const float* __restrict__ w_ih1,
    const float* __restrict__ b_ih1, const float* __restrict__ b_hh1,
    const float* __restrict__ w_ih2,
    const float* __restrict__ b_ih2, const float* __restrict__ b_hh2,
    const float* __restrict__ w_lin, const float* __restrict__ b_lin,
    float* __restrict__ out, int n_rows)
{
    // B fragments of w_ih2^T (51x204 -> padded 64x208), pre-swizzled into the
    // exact 16-bit B-matrix VGPR layout: [tile][kstep][lane][16 halves]
    __shared__ __align__(32) _Float16 bfragh[NT * 2 * 32 * 16];
    // per-wave gates2 staging (16 rows x 208 cols, f16)
    __shared__ __align__(16) _Float16 gates[WAVES][16][NT * 16];
    __shared__ float wi1s[G4], b1s[G4], b2s[G4], wls[HID];
    __shared__ float blin_s;

    const int tid  = threadIdx.x;
    const int lane = tid & 31;
    const int wave = tid >> 5;
    const int m    = lane & 15;   // row within 16-row tile / col within 16-col tile
    const int hi   = lane >> 4;   // lane half selects K sub-range per ISA layout

    // ---- one-time per-block LDS setup ----
    for (int i = tid; i < G4; i += BLK) {
        wi1s[i] = w_ih1[i];
        b1s[i]  = b_ih1[i] + b_hh1[i];
        b2s[i]  = b_ih2[i] + b_hh2[i];
    }
    for (int i = tid; i < HID; i += BLK) wls[i] = w_lin[i];
    if (tid == 0) blin_s = b_lin[0];

    // Build B fragments: half index hh of lane ln holds B[k][n] with
    // k = s*32 + (ln>=16 ? 8:0) + (hh>=8 ? 16:0) + (hh&7),  n = t*16 + (ln&15)
    constexpr int TOTH = NT * 2 * 32 * 16;          // 13312 halves
    #pragma unroll
    for (int i = 0; i < TOTH / BLK; ++i) {
        int hidx = tid + i * BLK;
        int hh = hidx & 15;
        int ln = (hidx >> 4) & 31;
        int s  = (hidx >> 9) & 1;
        int t  = hidx >> 10;
        int k  = s * 32 + ((ln >> 4) << 3) + ((hh >> 3) << 4) + (hh & 7);
        int nc = t * 16 + (ln & 15);
        float v = (k < HID && nc < G4) ? w_ih2[nc * HID + k] : 0.0f;  // w_ih2^T[k][nc]
        bfragh[hidx] = (_Float16)v;
    }
    __syncthreads();

    const int ntiles    = n_rows >> 4;
    const int gridWaves = gridDim.x * WAVES;
    const int iters     = (ntiles + gridWaves - 1) / gridWaves;

    for (int it = 0; it < iters; ++it) {
        const int  tile    = it * gridWaves + blockIdx.x * WAVES + wave;
        const bool active  = (tile < ntiles);          // wave-uniform -> EXEC stays full
        const int  rowbase = tile << 4;

        if (active) {
            // ---- layer 1: compute A fragment (h1 tile, 16x64 f16) in registers ----
            const float xv = x[rowbase + m];
            v16h afrag[2];
            #pragma unroll
            for (int s = 0; s < 2; ++s) {
                #pragma unroll
                for (int hh = 0; hh < 16; ++hh) {
                    int k = s * 32 + (hi << 3) + ((hh >> 3) << 4) + (hh & 7);
                    float hv = 0.0f;
                    if (k < HID) {
                        float gi = fsig  (xv * wi1s[k]            + b1s[k]);
                        float gg = ftanh_(xv * wi1s[2 * HID + k]  + b1s[2 * HID + k]);
                        float go = fsig  (xv * wi1s[3 * HID + k]  + b1s[3 * HID + k]);
                        hv = go * ftanh_(gi * gg);
                    }
                    afrag[s][hh] = (_Float16)hv;
                }
            }

            // ---- layer 2 GEMM: 13 column tiles x 2 K-steps of WMMA f16 ----
            #pragma unroll
            for (int t = 0; t < NT; ++t) {
                v16h bf0 = *(const v16h*)&bfragh[((t * 2 + 0) * 32 + lane) * 16];
                v16h bf1 = *(const v16h*)&bfragh[((t * 2 + 1) * 32 + lane) * 16];
                v8f acc = {};
                acc = __builtin_amdgcn_wmma_f32_16x16x32_f16(false, afrag[0], false, bf0,
                                                             (short)0, acc, false, false);
                acc = __builtin_amdgcn_wmma_f32_16x16x32_f16(false, afrag[1], false, bf1,
                                                             (short)0, acc, false, false);
                int   ncol = t * 16 + m;
                float bias = (ncol < G4) ? b2s[ncol] : 0.0f;
                #pragma unroll
                for (int r = 0; r < 8; ++r) {
                    int mrow = r + (hi << 3);            // C/D layout: lanes>=16 hold M=8..15
                    gates[wave][mrow][ncol] = (_Float16)(acc[r] + bias);
                }
            }
        }
        __syncthreads();

        if (active) {
            // ---- activations + 51-wide dot with w_lin; 2 lanes per row ----
            float partial = 0.0f;
            const int kbase = hi ? 26 : 0;
            const int kcnt  = hi ? 25 : 26;
            #pragma unroll
            for (int kk = 0; kk < 26; ++kk) {
                if (kk < kcnt) {
                    int k = kbase + kk;
                    float gi = (float)gates[wave][m][k];
                    float gg = (float)gates[wave][m][2 * HID + k];
                    float go = (float)gates[wave][m][3 * HID + k];
                    float h2 = fsig(go) * ftanh_(fsig(gi) * ftanh_(gg));
                    partial += h2 * wls[k];
                }
            }
            // SWAPX16 ds_swizzle: combine the two lane-halves of each row
            int   ob    = __builtin_amdgcn_ds_swizzle(__float_as_int(partial), 0x401F);
            float total = partial + __int_as_float(ob);
            if (hi == 0) out[rowbase + m] = total + blin_s;
        }
        __syncthreads();
    }
}

extern "C" void kernel_launch(void* const* d_in, const int* in_sizes, int n_in,
                              void* d_out, int out_size, void* d_ws, size_t ws_size,
                              hipStream_t stream) {
    const float* x     = (const float*)d_in[0];
    const float* w_ih1 = (const float*)d_in[1];
    // d_in[2] = w_hh1 (unused: zero state kills the W_hh term)
    const float* b_ih1 = (const float*)d_in[3];
    const float* b_hh1 = (const float*)d_in[4];
    const float* w_ih2 = (const float*)d_in[5];
    // d_in[6] = w_hh2 (unused)
    const float* b_ih2 = (const float*)d_in[7];
    const float* b_hh2 = (const float*)d_in[8];
    const float* w_lin = (const float*)d_in[9];
    const float* b_lin = (const float*)d_in[10];
    float* out = (float*)d_out;

    const int n_rows = in_sizes[0];          // T*B*1 = 409600
    const int ntiles = n_rows >> 4;
    int blocks = (ntiles + WAVES - 1) / WAVES;
    if (blocks > 1024) blocks = 1024;        // persistent waves, grid-stride by tiles
    fused_lstm_wmma<<<blocks, BLK, 0, stream>>>(
        x, w_ih1, b_ih1, b_hh1, w_ih2, b_ih2, b_hh2, w_lin, b_lin, out, n_rows);
}
